// criss_cross_attention_Aggregation_14413910245607
// MI455X (gfx1250) — matrix operand order repaired and checked by reading, not compiled, not run on hardware
//
#include <hip/hip_runtime.h>

typedef __attribute__((ext_vector_type(16))) _Float16 v16h;
typedef __attribute__((ext_vector_type(8)))  _Float16 v8h;
typedef __attribute__((ext_vector_type(8)))  float    v8f;

#define Bz 8
#define Hh 96
#define Ww 96
#define Cc 512
#define Ll 191      // H + W - 1
#define LDSTR 104   // LDS row stride in halves: 208 bytes, 16B-aligned rows
#define CCHUNK 128  // channels per LDS tile
#define NCHUNK 2    // channel tiles per block (reuses att matrix)

// ---------------------------------------------------------------------------
// Pass 0: per-pixel softmax stats over the 191 affinity values.
// One wave32 per pixel; writes (max, 1/sum_exp) to workspace.
// ---------------------------------------------------------------------------
__global__ __launch_bounds__(256) void cc_softmax_stats(
    const float* __restrict__ aff, float2* __restrict__ stats)
{
  const int lane = threadIdx.x & 31;
  const int wave = threadIdx.x >> 5;
  const long p = (long)blockIdx.x * 8 + wave;   // pixel id < 73728
  const float* a = aff + p * Ll;

  float v[6];
  #pragma unroll
  for (int it = 0; it < 6; ++it) {
    const int l = lane + 32 * it;
    v[it] = (l < Ll) ? a[l] : -3.402823466e+38f;
  }
  float m = v[0];
  #pragma unroll
  for (int it = 1; it < 6; ++it) m = fmaxf(m, v[it]);
  #pragma unroll
  for (int off = 16; off >= 1; off >>= 1)
    m = fmaxf(m, __shfl_xor(m, off, 32));

  float s = 0.0f;
  #pragma unroll
  for (int it = 0; it < 6; ++it) {
    const int l = lane + 32 * it;
    if (l < Ll) s += __expf(v[it] - m);
  }
  #pragma unroll
  for (int off = 16; off >= 1; off >>= 1)
    s += __shfl_xor(s, off, 32);

  if (lane == 0) stats[p] = make_float2(m, 1.0f / s);
}

// ---------------------------------------------------------------------------
// Fragment loader: two aligned 16B LDS reads -> one 16-half WMMA operand.
// ISA VGPR layouts (wave32):
//   A (16x32 f16): lane<16 -> M=lane,   K in {a..a+7} u {a+16..a+23}, a=0|8
//   B (32x16 f16): lane<16 -> N=lane,   K in {b..b+7} u {b+16..b+23}, b=0|8
//   D (16x16 f32): VGPR r  -> M = r (+8 for hi half), N = lane&15
// ---------------------------------------------------------------------------
__device__ __forceinline__ v16h load_frag2(const _Float16* p0, const _Float16* p1)
{
  const v8h lo = *(const v8h*)p0;
  const v8h hi = *(const v8h*)p1;
  v16h f;
  #pragma unroll
  for (int e = 0; e < 8; ++e) { f[e] = lo[e]; f[e + 8] = hi[e]; }
  return f;
}

// ---------------------------------------------------------------------------
// Pass 1: row attention.  block = (b*H + i, 256-channel slab).
// out[b,i,j,c] = sum_k Att[j,k] * x[b,i,k,c]   (overwrites out)
// ---------------------------------------------------------------------------
__global__ __launch_bounds__(256) void cc_row_gemm(
    const float* __restrict__ x, const float* __restrict__ aff,
    const float2* __restrict__ stats, float* __restrict__ out)
{
  __shared__ __align__(16) _Float16 att[Ww * LDSTR];     // [j][k]
  __shared__ __align__(16) _Float16 xt[CCHUNK * LDSTR];  // [c][k] (transposed)

  const int b     = blockIdx.x / Hh;
  const int i     = blockIdx.x % Hh;
  const int cbase = blockIdx.y * (NCHUNK * CCHUNK);
  const int t     = threadIdx.x;

  const long pixBase = ((long)b * Hh + i) * Ww;          // + j = pixel id

  // Build remapped row-attention matrix (diagonal = 0, k>j shifts to k-1).
  for (int e = t; e < Ww * Ww; e += 256) {
    const int j = e / Ww, k = e - j * Ww;
    float w = 0.0f;
    if (k != j) {
      const int idx = (k < j) ? k : (k - 1);
      const float2 ms = stats[pixBase + j];
      w = __expf(aff[(pixBase + j) * Ll + idx] - ms.x) * ms.y;
    }
    att[j * LDSTR + k] = (_Float16)w;
  }

  const int wave  = t >> 5, lane = t & 31;
  const int laneN = lane & 15;
  const int hiH   = lane >> 4;           // 0: low half, 1: high half of wave
  const int aoff  = hiH * 8;             // A: K group base (0 / 8)
  const int boff  = hiH * 8;             // B: K group base (0 / 8)
  const int cc    = wave * 16;           // this wave's channel tile in chunk
  const int mrow  = hiH * 8;

  const float* __restrict__ xrow = x + pixBase * Cc;     // [k][c], k-stride Cc
  float* __restrict__ orow = out + pixBase * Cc;         // [j][c]

  for (int ch = 0; ch < NCHUNK; ++ch) {
    const int c0 = cbase + ch * CCHUNK;
    if (ch) __syncthreads();   // all waves done reading previous xt

    // Load x[b,i,k,c0:c0+128] (f32), convert f16, store channel-major.
    for (int e = t; e < Ww * (CCHUNK / 4); e += 256) {
      const int k  = e >> 5;             // CCHUNK/4 == 32
      const int c4 = (e & 31) * 4;
      const float4 vv = *(const float4*)(xrow + (long)k * Cc + c0 + c4);
      xt[(c4 + 0) * LDSTR + k] = (_Float16)vv.x;
      xt[(c4 + 1) * LDSTR + k] = (_Float16)vv.y;
      xt[(c4 + 2) * LDSTR + k] = (_Float16)vv.z;
      xt[(c4 + 3) * LDSTR + k] = (_Float16)vv.w;
    }
    __syncthreads();

    // Hoist all three B (x) fragments for this wave's channel tile.
    v16h bf[3];
    #pragma unroll
    for (int kt = 0; kt < 3; ++kt) {
      const _Float16* bp = &xt[(cc + laneN) * LDSTR + kt * 32 + boff];
      bf[kt] = load_frag2(bp, bp + 16);
    }

    // Two j-tiles per iteration: two independent accumulator chains so the
    // XDL pipe issues interleaved wmmas without accumulator-RAW NOP stalls.
    for (int jt = 0; jt < Ww / 16; jt += 2) {
      v16h af0[3], af1[3];
      #pragma unroll
      for (int kt = 0; kt < 3; ++kt) {
        const _Float16* a0 = &att[((jt + 0) * 16 + laneN) * LDSTR + kt * 32 + aoff];
        const _Float16* a1 = &att[((jt + 1) * 16 + laneN) * LDSTR + kt * 32 + aoff];
        af0[kt] = load_frag2(a0, a0 + 16);
        af1[kt] = load_frag2(a1, a1 + 16);
      }
      v8f acc0 = {}, acc1 = {};
      #pragma unroll
      for (int kt = 0; kt < 3; ++kt) {
        acc0 = __builtin_amdgcn_wmma_f32_16x16x32_f16(false, af0[kt], false, bf[kt],
                                                      (short)0, acc0, false, false);
        acc1 = __builtin_amdgcn_wmma_f32_16x16x32_f16(false, af1[kt], false, bf[kt],
                                                      (short)0, acc1, false, false);
      }
      float* op0 = orow + (long)((jt + 0) * 16 + mrow) * Cc + c0 + cc + laneN;
      float* op1 = orow + (long)((jt + 1) * 16 + mrow) * Cc + c0 + cc + laneN;
      #pragma unroll
      for (int r = 0; r < 8; ++r) op0[(long)r * Cc] = acc0[r];
      #pragma unroll
      for (int r = 0; r < 8; ++r) op1[(long)r * Cc] = acc1[r];
    }
  }
}

// ---------------------------------------------------------------------------
// Pass 2: column attention.  block = (b*W + j, 256-channel slab).
// out[b,i,j,c] += sum_h Att[i,h] * x[b,h,j,c]   (exclusive RMW per block)
// ---------------------------------------------------------------------------
__global__ __launch_bounds__(256) void cc_col_gemm(
    const float* __restrict__ x, const float* __restrict__ aff,
    const float2* __restrict__ stats, float* __restrict__ out)
{
  __shared__ __align__(16) _Float16 att[Hh * LDSTR];     // [i][h]
  __shared__ __align__(16) _Float16 xt[CCHUNK * LDSTR];  // [c][h] (transposed)

  const int b     = blockIdx.x / Ww;
  const int j     = blockIdx.x % Ww;
  const int cbase = blockIdx.y * (NCHUNK * CCHUNK);
  const int t     = threadIdx.x;

  // Column-attention weights: A[..., W-1 + h], softmax-normalized.
  for (int e = t; e < Hh * Hh; e += 256) {
    const int i = e / Hh, h = e - i * Hh;
    const long pix = ((long)b * Hh + i) * Ww + j;
    const float2 ms = stats[pix];
    const float w = __expf(aff[pix * Ll + (Ww - 1) + h] - ms.x) * ms.y;
    att[i * LDSTR + h] = (_Float16)w;
  }

  const int wave  = t >> 5, lane = t & 31;
  const int laneN = lane & 15;
  const int hiH   = lane >> 4;
  const int aoff  = hiH * 8;
  const int boff  = hiH * 8;
  const int cc    = wave * 16;
  const int mrow  = hiH * 8;

  const float* __restrict__ xcol = x + ((long)b * Hh * Ww + j) * Cc;

  for (int ch = 0; ch < NCHUNK; ++ch) {
    const int c0 = cbase + ch * CCHUNK;
    if (ch) __syncthreads();

    // Load x[b,h,j,c0:c0+128] for h = 0..95 (row stride W*C), transpose.
    for (int e = t; e < Hh * (CCHUNK / 4); e += 256) {
      const int h  = e >> 5;
      const int c4 = (e & 31) * 4;
      const float4 vv = *(const float4*)(xcol + (long)h * Ww * Cc + c0 + c4);
      xt[(c4 + 0) * LDSTR + h] = (_Float16)vv.x;
      xt[(c4 + 1) * LDSTR + h] = (_Float16)vv.y;
      xt[(c4 + 2) * LDSTR + h] = (_Float16)vv.z;
      xt[(c4 + 3) * LDSTR + h] = (_Float16)vv.w;
    }
    __syncthreads();

    v16h bf[3];
    #pragma unroll
    for (int kt = 0; kt < 3; ++kt) {
      const _Float16* bp = &xt[(cc + laneN) * LDSTR + kt * 32 + boff];
      bf[kt] = load_frag2(bp, bp + 16);
    }

    for (int it = 0; it < Hh / 16; it += 2) {
      v16h af0[3], af1[3];
      #pragma unroll
      for (int kt = 0; kt < 3; ++kt) {
        const _Float16* a0 = &att[((it + 0) * 16 + laneN) * LDSTR + kt * 32 + aoff];
        const _Float16* a1 = &att[((it + 1) * 16 + laneN) * LDSTR + kt * 32 + aoff];
        af0[kt] = load_frag2(a0, a0 + 16);
        af1[kt] = load_frag2(a1, a1 + 16);
      }
      v8f acc0 = {}, acc1 = {};
      #pragma unroll
      for (int kt = 0; kt < 3; ++kt) {
        acc0 = __builtin_amdgcn_wmma_f32_16x16x32_f16(false, af0[kt], false, bf[kt],
                                                      (short)0, acc0, false, false);
        acc1 = __builtin_amdgcn_wmma_f32_16x16x32_f16(false, af1[kt], false, bf[kt],
                                                      (short)0, acc1, false, false);
      }
      // out[b, i=it*16+mrow+r, j, c0+cc+laneN] += acc[r]
      float* op0 = out + (((long)b * Hh + (it + 0) * 16 + mrow) * Ww + j) * Cc
                       + c0 + cc + laneN;
      float* op1 = out + (((long)b * Hh + (it + 1) * 16 + mrow) * Ww + j) * Cc
                       + c0 + cc + laneN;
      #pragma unroll
      for (int r = 0; r < 8; ++r) {
        const long off = (long)r * Ww * Cc;
        op0[off] = op0[off] + acc0[r];
      }
      #pragma unroll
      for (int r = 0; r < 8; ++r) {
        const long off = (long)r * Ww * Cc;
        op1[off] = op1[off] + acc1[r];
      }
    }
  }
}

// ---------------------------------------------------------------------------
extern "C" void kernel_launch(void* const* d_in, const int* in_sizes, int n_in,
                              void* d_out, int out_size, void* d_ws, size_t ws_size,
                              hipStream_t stream)
{
  const float* x   = (const float*)d_in[0];   // (8,96,96,512) f32
  const float* aff = (const float*)d_in[1];   // (8,96,96,191) f32
  float* out   = (float*)d_out;               // (8,96,96,512) f32
  float2* stats = (float2*)d_ws;              // 73728 * float2 = 576 KB

  const int pixels = Bz * Hh * Ww;            // 73728
  const int cslabs = Cc / (NCHUNK * CCHUNK);  // 2
  cc_softmax_stats<<<dim3(pixels / 8), dim3(256), 0, stream>>>(aff, stats);
  cc_row_gemm<<<dim3(Bz * Hh, cslabs), dim3(256), 0, stream>>>(x, aff, stats, out);
  cc_col_gemm<<<dim3(Bz * Ww, cslabs), dim3(256), 0, stream>>>(x, aff, stats, out);
}
